// TitansBlock_62448824483953
// MI455X (gfx1250) — compile-verified, matrix operand order checked by reading, not run on hardware
//
#include <hip/hip_runtime.h>
#include <math.h>

#define S_     16384
#define D_     1024
#define T_     1024
#define LC_    77
#define H_     8
#define DH_    64
#define CH_    128
#define INNER_ 512
#define NCH_   128
#define FF_    4096

typedef unsigned short u16;
typedef __bf16 v16bf __attribute__((ext_vector_type(16)));
typedef float  v8f   __attribute__((ext_vector_type(8)));

__device__ __forceinline__ u16 f2bf(float f) {
    unsigned u = __float_as_uint(f);
    unsigned r = 0x7FFFu + ((u >> 16) & 1u);
    return (u16)((u + r) >> 16);
}
__device__ __forceinline__ float bf2f(u16 h) {
    return __uint_as_float(((unsigned)h) << 16);
}
__device__ __forceinline__ float gelu_exact(float x) {
    return 0.5f * x * (1.0f + erff(x * 0.70710678118654752f));
}
__device__ __forceinline__ float sigmoidf_(float x) {
    return 1.0f / (1.0f + expf(-x));
}

// ---------------------------------------------------------------------------
// f32 -> bf16 conversion (weights / activations)
// ---------------------------------------------------------------------------
__global__ __launch_bounds__(256) void k_tobf(const float* __restrict__ src,
                                              u16* __restrict__ dst, size_t n) {
    size_t i = (size_t)blockIdx.x * 256 + threadIdx.x;
    if (i < n) dst[i] = f2bf(src[i]);
}

// ---------------------------------------------------------------------------
// out[j] = b[j] + sum_t te[t] * W[t*ncols + j]   (modulation / cond-gate vecs)
// ---------------------------------------------------------------------------
__global__ __launch_bounds__(256) void k_modvec(const float* __restrict__ te,
                                                const float* __restrict__ W,
                                                const float* __restrict__ b,
                                                float* __restrict__ out, int ncols) {
    int j = blockIdx.x * 256 + threadIdx.x;
    if (j >= ncols) return;
    float a = b[j];
    for (int t = 0; t < T_; ++t) a += te[t] * W[(size_t)t * ncols + j];
    out[j] = a;
}

// ---------------------------------------------------------------------------
// LayerNorm + dual modulation: x_mixer (f32 + bf16), x_ff (bf16)
// ---------------------------------------------------------------------------
__global__ __launch_bounds__(256) void k_ln_mod(const float* __restrict__ x,
                                                const float* __restrict__ m,
                                                float* __restrict__ xm,
                                                u16* __restrict__ xmb,
                                                u16* __restrict__ xfb) {
    int s = blockIdx.x;
    const float* row = x + (size_t)s * D_;
    float v[4], lsum = 0.f, lsq = 0.f;
    for (int i = 0; i < 4; ++i) {
        float t = row[threadIdx.x + 256 * i];
        v[i] = t; lsum += t; lsq += t * t;
    }
    __shared__ float r1[256], r2[256];
    r1[threadIdx.x] = lsum; r2[threadIdx.x] = lsq;
    __syncthreads();
    for (int o = 128; o > 0; o >>= 1) {
        if (threadIdx.x < o) { r1[threadIdx.x] += r1[threadIdx.x + o]; r2[threadIdx.x] += r2[threadIdx.x + o]; }
        __syncthreads();
    }
    float mu = r1[0] * (1.0f / D_);
    float var = r2[0] * (1.0f / D_) - mu * mu;
    float rs = rsqrtf(var + 1e-5f);
    for (int i = 0; i < 4; ++i) {
        int d = threadIdx.x + 256 * i;
        float xn = (v[i] - mu) * rs;
        float a = xn * (1.f + m[D_ + d]) + m[d];
        float b = xn * (1.f + m[4 * D_ + d]) + m[3 * D_ + d];
        size_t idx = (size_t)s * D_ + d;
        xm[idx] = a; xmb[idx] = f2bf(a); xfb[idx] = f2bf(b);
    }
}

// LayerNorm only -> bf16 (for attention query path)
__global__ __launch_bounds__(256) void k_ln_plain(const float* __restrict__ x,
                                                  u16* __restrict__ ob) {
    int s = blockIdx.x;
    const float* row = x + (size_t)s * D_;
    float v[4], lsum = 0.f, lsq = 0.f;
    for (int i = 0; i < 4; ++i) {
        float t = row[threadIdx.x + 256 * i];
        v[i] = t; lsum += t; lsq += t * t;
    }
    __shared__ float r1[256], r2[256];
    r1[threadIdx.x] = lsum; r2[threadIdx.x] = lsq;
    __syncthreads();
    for (int o = 128; o > 0; o >>= 1) {
        if (threadIdx.x < o) { r1[threadIdx.x] += r1[threadIdx.x + o]; r2[threadIdx.x] += r2[threadIdx.x + o]; }
        __syncthreads();
    }
    float mu = r1[0] * (1.0f / D_);
    float var = r2[0] * (1.0f / D_) - mu * mu;
    float rs = rsqrtf(var + 1e-5f);
    for (int i = 0; i < 4; ++i) {
        int d = threadIdx.x + 256 * i;
        ob[(size_t)s * D_ + d] = f2bf((v[i] - mu) * rs);
    }
}

// ---------------------------------------------------------------------------
// Depthwise 3x3x3 conv (SAME) + gelu + residual -> concat[:, 0:1024] (bf16)
// ---------------------------------------------------------------------------
__global__ __launch_bounds__(256) void k_conv(const float* __restrict__ xm,
                                              const float* __restrict__ w,
                                              const float* __restrict__ cb,
                                              u16* __restrict__ catb) {
    size_t idx = (size_t)blockIdx.x * 256 + threadIdx.x;
    int d = (int)(idx & (D_ - 1));
    int s = (int)(idx >> 10);
    int t = s >> 10, hh = (s >> 5) & 31, ww = s & 31;
    float acc = cb[d];
    const float* wd = w + (size_t)d * 27;
    for (int dz = 0; dz < 3; ++dz) {
        int tz = t + dz - 1; if (tz < 0 || tz >= 16) continue;
        for (int dy = 0; dy < 3; ++dy) {
            int hy = hh + dy - 1; if (hy < 0 || hy >= 32) continue;
            for (int dx = 0; dx < 3; ++dx) {
                int wx = ww + dx - 1; if (wx < 0 || wx >= 32) continue;
                int sp = (tz << 10) + (hy << 5) + wx;
                acc += wd[dz * 9 + dy * 3 + dx] * xm[(size_t)sp * D_ + d];
            }
        }
    }
    float lo = xm[idx] + gelu_exact(acc);
    catb[(size_t)s * (2 * D_) + d] = f2bf(lo);
}

// ---------------------------------------------------------------------------
// Per-chunk mean of sigmoid(x @ Wg + bg)  ->  gout[n*24 + j]
// ---------------------------------------------------------------------------
__global__ __launch_bounds__(256) void k_gproj(const float* __restrict__ xm,
                                               const float* __restrict__ Wg,
                                               const float* __restrict__ bg,
                                               float* __restrict__ gout, int rev) {
    __shared__ float accs[24];
    int n = blockIdx.x, tid = threadIdx.x;
    if (tid < 24) accs[tid] = 0.f;
    __syncthreads();
    for (int pi = tid; pi < CH_ * 24; pi += 256) {
        int l = pi / 24, j = pi - l * 24;
        int p = n * CH_ + l;
        int s = rev ? (S_ - 1 - p) : p;
        const float* xr = xm + (size_t)s * D_;
        float a = bg[j];
        for (int d = 0; d < D_; ++d) a += xr[d] * Wg[(size_t)d * 24 + j];
        atomicAdd(&accs[j], sigmoidf_(a) * (1.0f / CH_));
    }
    __syncthreads();
    if (tid < 24) gout[n * 24 + tid] = accs[tid];
}

// ---------------------------------------------------------------------------
// Neural-memory scan (sequential over 128 chunks). One block per head.
// M, S state (64x64 f32) in LDS; pred-v chunk cached as bf16 in LDS.
// ---------------------------------------------------------------------------
__global__ __launch_bounds__(256) void k_memscan(const float* __restrict__ q,
                                                 const float* __restrict__ kk,
                                                 const float* __restrict__ vv,
                                                 const float* __restrict__ g,
                                                 u16* __restrict__ ysb, int rev) {
    __shared__ float Ms[64 * 64];
    __shared__ float Ss[64 * 64];
    __shared__ u16  pvs[CH_ * 64];
    int h = blockIdx.x, tid = threadIdx.x;
    for (int i = tid; i < 4096; i += 256) { Ms[i] = 0.f; Ss[i] = 0.f; }
    __syncthreads();
    for (int n = 0; n < NCH_; ++n) {
        {   // phase A: y = q@M, pv = k@M - v
            int l = tid >> 1;
            int eo = (tid & 1) * 32;
            int p = n * CH_ + l;
            int s = rev ? (S_ - 1 - p) : p;
            const float* qr = q  + (size_t)s * INNER_ + h * 64;
            const float* kr = kk + (size_t)s * INNER_ + h * 64;
            const float* vr = vv + (size_t)s * INNER_ + h * 64;
            float ya[32], pa[32];
            #pragma unroll
            for (int e = 0; e < 32; ++e) { ya[e] = 0.f; pa[e] = 0.f; }
            for (int d = 0; d < 64; ++d) {
                float qd = qr[d], kd = kr[d];
                const float* Mrow = Ms + d * 64 + eo;
                #pragma unroll
                for (int e = 0; e < 32; ++e) { ya[e] += qd * Mrow[e]; pa[e] += kd * Mrow[e]; }
            }
            u16* yo = ysb + (size_t)s * INNER_ + h * 64 + eo;
            #pragma unroll
            for (int e = 0; e < 32; ++e) {
                yo[e] = f2bf(ya[e]);
                pvs[l * 64 + eo + e] = f2bf(pa[e] - vr[eo + e]);
            }
        }
        __syncthreads();
        {   // phase B: grad = k^T @ pv / CH ; S,M update
            int d = tid >> 2;
            int eo = (tid & 3) * 16;
            float ga[16];
            #pragma unroll
            for (int e = 0; e < 16; ++e) ga[e] = 0.f;
            for (int l = 0; l < CH_; ++l) {
                int p = n * CH_ + l;
                int s = rev ? (S_ - 1 - p) : p;
                float kd = kk[(size_t)s * INNER_ + h * 64 + d];
                const u16* pr = pvs + l * 64 + eo;
                #pragma unroll
                for (int e = 0; e < 16; ++e) ga[e] += kd * bf2f(pr[e]);
            }
            float al = g[n * 24 + h];
            float et = g[n * 24 + 8 + h];
            float th = g[n * 24 + 16 + h];
            #pragma unroll
            for (int e = 0; e < 16; ++e) {
                int idx = d * 64 + eo + e;
                float sm = et * Ss[idx] - th * (ga[e] * (1.0f / CH_));
                Ss[idx] = sm;
                Ms[idx] = (1.f - al) * Ms[idx] + sm;
            }
        }
        __syncthreads();
    }
}

// ---------------------------------------------------------------------------
// WMMA bf16 GEMM: C[M,N] = act(alpha * A[M,K] @ B[K,N] + bias (+C if accum))
// 128x128 block tile, BK=32, 8 waves, each wave a 16x128 strip (8 wmma tiles).
// Both A and B tiles are staged in LDS in fragment-major ([row][k]) order so
// every WMMA operand is fetched with two aligned ds_load_b128 per matrix.
// ---------------------------------------------------------------------------
__global__ __launch_bounds__(256) void k_gemm(const u16* __restrict__ A,
                                              const u16* __restrict__ B,
                                              const float* __restrict__ bias,
                                              float* __restrict__ C,
                                              u16* __restrict__ Cb,
                                              int M, int N, int K,
                                              float alpha, int act, int accum) {
    __shared__ u16 As[128 * 32];   // [m][k]
    __shared__ u16 Bs[128 * 32];   // [n][k]  (transposed at staging time)
    int tid = threadIdx.x;
    int lane = tid & 31;
    int wave = tid >> 5;
    int bm = blockIdx.y * 128;
    int bn = blockIdx.x * 128;

    v8f acc[8];
    #pragma unroll
    for (int t = 0; t < 8; ++t)
        #pragma unroll
        for (int r = 0; r < 8; ++r) acc[t][r] = 0.f;

    int am = (wave << 4) + (lane & 15);
    int kb = (lane >> 4) << 3;       // 0 or 8

    // staging decomposition (per thread)
    const int ar = tid >> 1;               // A row 0..127
    const int ac = (tid & 1) << 4;         // A col 0 or 16
    const int brp = tid >> 4;              // B row-pair 0..15 -> rows 2brp, 2brp+1
    const int bcs = (tid & 15) << 3;       // B col seg: 8 cols

    for (int kt = 0; kt < K; kt += 32) {
        // prefetch next K-step tiles (global_prefetch_b8 on gfx1250)
        if (kt + 32 < K) {
            __builtin_prefetch((const void*)(A + (size_t)(bm + ar) * K + (kt + 32) + ac), 0, 1);
            __builtin_prefetch((const void*)(B + (size_t)(kt + 32 + 2 * brp) * N + bn + bcs), 0, 1);
        }
        {   // stage A tile 128x32: two 16B loads + two 16B LDS stores
            const uint4* src = reinterpret_cast<const uint4*>(A + (size_t)(bm + ar) * K + kt + ac);
            uint4* dst = reinterpret_cast<uint4*>(&As[ar * 32 + ac]);
            dst[0] = src[0]; dst[1] = src[1];
        }
        {   // stage B tile 32x128 transposed into [n][k]
            const uint4 r0 = *reinterpret_cast<const uint4*>(B + (size_t)(kt + 2 * brp) * N + bn + bcs);
            const uint4 r1 = *reinterpret_cast<const uint4*>(B + (size_t)(kt + 2 * brp + 1) * N + bn + bcs);
            const u16* e0 = reinterpret_cast<const u16*>(&r0);
            const u16* e1 = reinterpret_cast<const u16*>(&r1);
            #pragma unroll
            for (int i = 0; i < 8; ++i) {
                unsigned pack = (unsigned)e0[i] | ((unsigned)e1[i] << 16);
                *reinterpret_cast<unsigned*>(&Bs[(bcs + i) * 32 + 2 * brp]) = pack;
            }
        }
        __syncthreads();

        // A fragment: elements {kb..kb+7, kb+16..kb+23} of row am -> 2x b128
        union { v16bf v; uint4 q[2]; } af;
        {
            const uint4* s = reinterpret_cast<const uint4*>(&As[am * 32 + kb]);
            af.q[0] = s[0];
            af.q[1] = s[2];
        }
        #pragma unroll
        for (int t = 0; t < 8; ++t) {
            int bnl = (t << 4) + (lane & 15);
            union { v16bf v; uint4 q[2]; } bf;
            const uint4* s = reinterpret_cast<const uint4*>(&Bs[bnl * 32 + kb]);
            bf.q[0] = s[0];
            bf.q[1] = s[2];
            acc[t] = __builtin_amdgcn_wmma_f32_16x16x32_bf16(
                false, af.v, false, bf.v, (short)0, acc[t], false, false);
        }
        __syncthreads();
    }

    int rowBase = bm + (wave << 4) + ((lane >> 4) << 3);
    int colL = lane & 15;
    #pragma unroll
    for (int t = 0; t < 8; ++t) {
        int col = bn + (t << 4) + colL;
        float bv = bias ? bias[col] : 0.f;
        #pragma unroll
        for (int r = 0; r < 8; ++r) {
            size_t off = (size_t)(rowBase + r) * N + col;
            float x = acc[t][r] * alpha + bv;
            if (accum) x += C[off];
            if (act == 1)      x = gelu_exact(x);
            else if (act == 2) x = sigmoidf_(x);
            if (C)  C[off]  = x;
            if (Cb) Cb[off] = f2bf(x);
        }
    }
}

// ---------------------------------------------------------------------------
// cond K/V projections (77 rows, tiny)
// ---------------------------------------------------------------------------
__global__ __launch_bounds__(256) void k_condkv(const float* __restrict__ cond,
                                                const float* __restrict__ Wk,
                                                const float* __restrict__ bk,
                                                const float* __restrict__ Wv,
                                                const float* __restrict__ bv,
                                                float* __restrict__ kh,
                                                float* __restrict__ vh) {
    int idx = blockIdx.x * 256 + threadIdx.x;
    if (idx >= LC_ * D_) return;
    int i = idx >> 10, j = idx & (D_ - 1);
    const float* cr = cond + (size_t)i * D_;
    float ak = bk[j], av = bv[j];
    for (int d = 0; d < D_; ++d) {
        float c = cr[d];
        ak += c * Wk[(size_t)d * D_ + j];
        av += c * Wv[(size_t)d * D_ + j];
    }
    kh[idx] = ak; vh[idx] = av;
}

// ---------------------------------------------------------------------------
// Cross-attention over 77 keys. Block = 64 query rows of one head.
// ---------------------------------------------------------------------------
__global__ __launch_bounds__(64) void k_attn(const float* __restrict__ qf,
                                             const float* __restrict__ kh,
                                             const float* __restrict__ vh,
                                             u16* __restrict__ aob) {
    __shared__ float khs[LC_ * 64];
    __shared__ float vhs[LC_ * 64];
    __shared__ float sc[64 * 80];
    int h = blockIdx.y;
    int s = blockIdx.x * 64 + threadIdx.x;
    int tid = threadIdx.x;
    for (int i = tid; i < LC_ * 64; i += 64) {
        int ki = i >> 6, e = i & 63;
        khs[i] = kh[(size_t)ki * D_ + h * 64 + e];
        vhs[i] = vh[(size_t)ki * D_ + h * 64 + e];
    }
    __syncthreads();
    const float* qr = qf + (size_t)s * D_ + h * 64;
    float mx = -1e30f;
    for (int ki = 0; ki < LC_; ++ki) {
        float a = 0.f;
        for (int e = 0; e < 64; ++e) a += qr[e] * khs[ki * 64 + e];
        a *= 0.125f;
        sc[tid * 80 + ki] = a;
        mx = fmaxf(mx, a);
    }
    float se = 0.f;
    for (int ki = 0; ki < LC_; ++ki) {
        float p = expf(sc[tid * 80 + ki] - mx);
        sc[tid * 80 + ki] = p; se += p;
    }
    float inv = 1.f / se;
    u16* out = aob + (size_t)s * D_ + h * 64;
    for (int e = 0; e < 64; ++e) {
        float a = 0.f;
        for (int ki = 0; ki < LC_; ++ki) a += sc[tid * 80 + ki] * vhs[ki * 64 + e];
        out[e] = f2bf(a * inv);
    }
}

// ---------------------------------------------------------------------------
// Elementwise epilogues
// ---------------------------------------------------------------------------
__global__ __launch_bounds__(256) void k_pack_concat(const float* __restrict__ src,
                                                     u16* __restrict__ dst) {
    size_t idx = (size_t)blockIdx.x * 256 + threadIdx.x;
    int s = (int)(idx >> 10), d = (int)(idx & (D_ - 1));
    dst[(size_t)s * (2 * D_) + D_ + d] = f2bf(src[idx]);
}

__global__ __launch_bounds__(256) void k_combine(const float* __restrict__ tok,
                                                 const float* __restrict__ m,
                                                 const float* __restrict__ gate,
                                                 const float* __restrict__ mixed,
                                                 float* __restrict__ out) {
    size_t idx = (size_t)blockIdx.x * 256 + threadIdx.x;
    int d = (int)(idx & (D_ - 1));
    out[idx] = tok[idx] + m[2 * D_ + d] * mixed[idx] * gate[idx];
}

__global__ __launch_bounds__(256) void k_axpy_col(float* __restrict__ out,
                                                  const float* __restrict__ a,
                                                  const float* __restrict__ vec, int off) {
    size_t idx = (size_t)blockIdx.x * 256 + threadIdx.x;
    int d = (int)(idx & (D_ - 1));
    out[idx] += vec[off + d] * a[idx];
}

// ---------------------------------------------------------------------------
// Host side
// ---------------------------------------------------------------------------
extern "C" void kernel_launch(void* const* d_in, const int* in_sizes, int n_in,
                              void* d_out, int out_size, void* d_ws, size_t ws_size,
                              hipStream_t stream) {
    (void)in_sizes; (void)n_in; (void)out_size; (void)ws_size;

    const float* tokens  = (const float*)d_in[0];
    const float* t_embed = (const float*)d_in[1];
    const float* cond    = (const float*)d_in[2];
    const float* W_mod   = (const float*)d_in[3];
    const float* b_mod   = (const float*)d_in[4];
    const float* conv_w  = (const float*)d_in[5];
    const float* conv_b  = (const float*)d_in[6];
    const float* mem_Wq  = (const float*)d_in[7];
    const float* mem_Wk  = (const float*)d_in[8];
    const float* mem_Wv  = (const float*)d_in[9];
    const float* mem_Wo  = (const float*)d_in[10];
    const float* mem_Wg  = (const float*)d_in[11];
    const float* mem_bg  = (const float*)d_in[12];
    const float* gate_W  = (const float*)d_in[13];
    const float* gate_b  = (const float*)d_in[14];
    const float* mix_W   = (const float*)d_in[15];
    const float* mix_b   = (const float*)d_in[16];
    const float* ff_W1   = (const float*)d_in[17];
    const float* ff_b1   = (const float*)d_in[18];
    const float* ff_W2   = (const float*)d_in[19];
    const float* ff_b2   = (const float*)d_in[20];
    const float* attn_Wq = (const float*)d_in[21];
    const float* attn_Wk = (const float*)d_in[22];
    const float* attn_Wv = (const float*)d_in[23];
    const float* attn_bq = (const float*)d_in[24];
    const float* attn_bk = (const float*)d_in[25];
    const float* attn_bv = (const float*)d_in[26];
    const float* attn_Wo = (const float*)d_in[27];
    const float* attn_bo = (const float*)d_in[28];
    const float* cg_W    = (const float*)d_in[29];
    const float* cg_b    = (const float*)d_in[30];

    float* out = (float*)d_out;
    char* ws = (char*)d_ws;
    size_t off = 0;
    auto alloc = [&](size_t bytes) -> char* {
        char* p = ws + off;
        off += (bytes + 255) & ~(size_t)255;
        return p;
    };
    const size_t SZ  = (size_t)S_ * D_;
    const size_t SZI = (size_t)S_ * INNER_;

    float* xm   = (float*)alloc(SZ * 4);            // x_mixer f32
    float* tmp0 = (float*)alloc(SZ * 4);
    float* tmp1 = (float*)alloc(SZ * 4);
    float* gob  = (float*)alloc(SZ * 4);            // global_out f32 (later: ao bf16)
    u16*   xmb  = (u16*)alloc(SZ * 2);              // x_mixer bf16 (later: xq bf16)
    u16*   xfb  = (u16*)alloc(SZ * 2);              // x_ff bf16
    u16*   catb = (u16*)alloc((size_t)S_ * 2 * D_ * 2); // concat bf16
    char*  qkvB = alloc((size_t)S_ * FF_ * 2);      // q,k,v,ys region; later ffh bf16
    float* qb  = (float*)qkvB;
    float* kb  = qb + SZI;
    float* vb  = kb + SZI;
    u16*   ysb = (u16*)(vb + SZI);
    u16*   ffh = (u16*)qkvB;
    float* mvec = (float*)alloc(6 * D_ * 4);
    float* cg   = (float*)alloc(D_ * 4);
    float* gsc  = (float*)alloc(NCH_ * 24 * 4);
    float* khb  = (float*)alloc((size_t)LC_ * D_ * 4);
    float* vhb  = (float*)alloc((size_t)LC_ * D_ * 4);
    u16* wqB  = (u16*)alloc((size_t)2 * D_ * INNER_ * 2);
    u16* wkB  = (u16*)alloc((size_t)2 * D_ * INNER_ * 2);
    u16* wvB  = (u16*)alloc((size_t)2 * D_ * INNER_ * 2);
    u16* woB  = (u16*)alloc((size_t)2 * INNER_ * D_ * 2);
    u16* gwB  = (u16*)alloc((size_t)2 * D_ * D_ * 2);
    u16* mwB  = (u16*)alloc((size_t)2 * D_ * D_ * 2);
    u16* f1B  = (u16*)alloc((size_t)D_ * FF_ * 2);
    u16* f2B  = (u16*)alloc((size_t)FF_ * D_ * 2);
    u16* aqB  = (u16*)alloc((size_t)D_ * D_ * 2);
    u16* aoB  = (u16*)alloc((size_t)D_ * D_ * 2);
    u16* aobf = (u16*)gob;   // reuse global_out region for attention output (bf16)
    u16* xqb  = xmb;         // reuse x_mixer bf16 region for xq bf16

    auto tobf = [&](const float* s, u16* d, size_t n) {
        k_tobf<<<(unsigned)((n + 255) / 256), 256, 0, stream>>>(s, d, n);
    };
    tobf(mem_Wq, wqB, (size_t)2 * D_ * INNER_);
    tobf(mem_Wk, wkB, (size_t)2 * D_ * INNER_);
    tobf(mem_Wv, wvB, (size_t)2 * D_ * INNER_);
    tobf(mem_Wo, woB, (size_t)2 * INNER_ * D_);
    tobf(gate_W, gwB, (size_t)2 * D_ * D_);
    tobf(mix_W,  mwB, (size_t)2 * D_ * D_);
    tobf(ff_W1,  f1B, (size_t)D_ * FF_);
    tobf(ff_W2,  f2B, (size_t)FF_ * D_);
    tobf(attn_Wq, aqB, (size_t)D_ * D_);
    tobf(attn_Wo, aoB, (size_t)D_ * D_);

    auto gemm = [&](const u16* A, const u16* B, const float* bias, float* C, u16* Cb,
                    int M, int N, int K, float alpha, int act, int accum) {
        dim3 grid(N / 128, M / 128);
        k_gemm<<<grid, 256, 0, stream>>>(A, B, bias, C, Cb, M, N, K, alpha, act, accum);
    };
    const unsigned GSZ = (unsigned)(SZ / 256);
    const float qscale = 0.125f; // DH^-0.5

    // 1) modulation vectors
    k_modvec<<<(6 * D_) / 256, 256, 0, stream>>>(t_embed, W_mod, b_mod, mvec, 6 * D_);
    k_modvec<<<D_ / 256, 256, 0, stream>>>(t_embed, cg_W, cg_b, cg, D_);

    // 2) LN + modulation
    k_ln_mod<<<S_, 256, 0, stream>>>(tokens, mvec, xm, xmb, xfb);

    // 3) depthwise conv mixer -> concat[:, :D]
    k_conv<<<GSZ, 256, 0, stream>>>(xm, conv_w, conv_b, catb);

    // 4) neural memory (global_out = x_mixer + feat_fwd + feat_bwd)
    hipMemcpyAsync(gob, xm, SZ * 4, hipMemcpyDeviceToDevice, stream);
    for (int dir = 0; dir < 2; ++dir) {
        size_t wo = (size_t)dir * D_ * INNER_;
        gemm(xmb, wqB + wo, nullptr, qb, nullptr, S_, INNER_, D_, qscale, 0, 0);
        gemm(xmb, wkB + wo, nullptr, kb, nullptr, S_, INNER_, D_, 1.0f, 0, 0);
        gemm(xmb, wvB + wo, nullptr, vb, nullptr, S_, INNER_, D_, 1.0f, 0, 0);
        k_gproj<<<NCH_, 256, 0, stream>>>(xm, mem_Wg + (size_t)dir * D_ * 24,
                                          mem_bg + dir * 24, gsc, dir);
        k_memscan<<<H_, 256, 0, stream>>>(qb, kb, vb, gsc, ysb, dir);
        gemm(ysb, woB + (size_t)dir * INNER_ * D_, nullptr, gob, nullptr,
             S_, D_, INNER_, 1.0f, 0, 1);
    }

    // 5) concat[:, D:2D]; gate/mix GEMMs; token update 1
    k_pack_concat<<<GSZ, 256, 0, stream>>>(gob, catb);
    gemm(catb, gwB, gate_b, tmp0, nullptr, S_, D_, 2 * D_, 1.0f, 2, 0); // sigmoid gate
    gemm(catb, mwB, mix_b, tmp1, nullptr, S_, D_, 2 * D_, 1.0f, 0, 0);  // mixed
    k_combine<<<GSZ, 256, 0, stream>>>(tokens, mvec, tmp0, tmp1, out);

    // 6) cross-attention
    k_ln_plain<<<S_, 256, 0, stream>>>(out, xqb);
    gemm(xqb, aqB, attn_bq, tmp0, nullptr, S_, D_, D_, 1.0f, 0, 0);     // Q
    k_condkv<<<(LC_ * D_ + 255) / 256, 256, 0, stream>>>(cond, attn_Wk, attn_bk,
                                                         attn_Wv, attn_bv, khb, vhb);
    {
        dim3 grid(S_ / 64, H_);
        k_attn<<<grid, 64, 0, stream>>>(tmp0, khb, vhb, aobf);
    }
    gemm(aobf, aoB, attn_bo, tmp1, nullptr, S_, D_, D_, 1.0f, 0, 0);    // out proj
    k_axpy_col<<<GSZ, 256, 0, stream>>>(out, tmp1, cg, 0);

    // 7) FFN
    gemm(xfb, f1B, ff_b1, nullptr, ffh, S_, FF_, D_, 1.0f, 1, 0);       // gelu -> bf16
    gemm(ffh, f2B, ff_b2, tmp0, nullptr, S_, D_, FF_, 1.0f, 0, 0);
    k_axpy_col<<<GSZ, 256, 0, stream>>>(out, tmp0, mvec, 5 * D_);
}
